// ICLAttention_67516885893228
// MI455X (gfx1250) — matrix operand order repaired and checked by reading, not compiled, not run on hardware
//
#include <hip/hip_runtime.h>
#include <hip/hip_bf16.h>

typedef __attribute__((ext_vector_type(16))) _Float16 v16h;
typedef __attribute__((ext_vector_type(8)))  _Float16 h8;
typedef __attribute__((ext_vector_type(8)))  float    v8f;
typedef __attribute__((ext_vector_type(4)))  float    f4v;

#define WMMA_F16(A,B,C) __builtin_amdgcn_wmma_f32_16x16x32_f16( \
    false, (A), false, (B), (short)0, (C), false, false)

static constexpr int Bn = 2, Sn = 2048, En = 1024, Dn = 1024, Hn = 16, HDn = 64;
static constexpr float SCALE_F = 0.03125f;   // 1/sqrt(1024)
static constexpr float NEG_BIG = -1.0e30f;

// ---------------------------------------------------------------------------
// Elementwise fp32 -> f16 pack (8 elements/thread, b128 in / b128 out).
// ---------------------------------------------------------------------------
__global__ __launch_bounds__(256)
void pack_f16(const float* __restrict__ src, _Float16* __restrict__ dst, int n)
{
    const int i = (blockIdx.x * blockDim.x + threadIdx.x) * 8;
    if (i >= n) return;
    f4v a = *(const f4v*)(src + i);
    f4v b = *(const f4v*)(src + i + 4);
    h8 o;
#pragma unroll
    for (int e = 0; e < 4; ++e) { o[e] = (_Float16)a[e]; o[4 + e] = (_Float16)b[e]; }
    *(h8*)(dst + i) = o;
}

// ---------------------------------------------------------------------------
// GEMM: Y[M,N] = X[M,K] @ W[N,K]^T, all-f16 operands, f32 WMMA accumulate.
// Block 256 thr = 8 waves as 4(M) x 2(N); wave tile 32(M) x 64(N).
// Block tile 128x128. 8 WMMAs per 32-wide K step, zero conversions in loop.
// ---------------------------------------------------------------------------
template <typename OutT>
__global__ __launch_bounds__(256)
void gemm_f16(const _Float16* __restrict__ X, const _Float16* __restrict__ W,
              OutT* __restrict__ Y, int M, int N, int K)
{
    const int lane = threadIdx.x & 31;
    const int wave = threadIdx.x >> 5;
    const int wm   = wave & 3;           // M-direction wave coord (0..3)
    const int wn   = wave >> 2;          // N-direction wave coord (0..1)
    const int hi   = lane >> 4;
    const int ln   = lane & 15;
    const int m0   = blockIdx.x * 128 + wm * 32;
    const int n0   = blockIdx.y * 128 + wn * 64;

    const _Float16* arow0 = X + (size_t)(m0 + ln) * K;
    const _Float16* arow1 = X + (size_t)(m0 + 16 + ln) * K;

    v8f acc[2][4] = {};

    for (int k = 0; k < K; k += 32) {
        // A fragments for rows m0+ln and m0+16+ln (16x32 f16 each)
        v16h a0, a1;
        {
            h8 x0 = *(const h8*)(arow0 + k + hi * 16);
            h8 x1 = *(const h8*)(arow0 + k + hi * 16 + 8);
            h8 y0 = *(const h8*)(arow1 + k + hi * 16);
            h8 y1 = *(const h8*)(arow1 + k + hi * 16 + 8);
#pragma unroll
            for (int e = 0; e < 8; ++e) {
                a0[e] = x0[e]; a0[8 + e] = x1[e];
                a1[e] = y0[e]; a1[8 + e] = y1[e];
            }
        }
        // 4 B fragments (32x16), each one contiguous 32B load; 8 WMMAs
#pragma unroll
        for (int t = 0; t < 4; ++t) {
            const _Float16* bp = W + (size_t)(n0 + t * 16 + ln) * K + k + hi * 16;
            v16h b = *(const v16h*)(bp);
            acc[0][t] = WMMA_F16(a0, b, acc[0][t]);
            acc[1][t] = WMMA_F16(a1, b, acc[1][t]);
        }
        // stream-ahead prefetch of the A rows (gfx1250 global_prefetch)
        __builtin_prefetch(arow0 + k + 512, 0, 1);
        __builtin_prefetch(arow1 + k + 512, 0, 1);
    }

    // store D: VGPR r of tile (mi,t) -> row m0+mi*16+hi*8+r, col n0+t*16+ln
#pragma unroll
    for (int mi = 0; mi < 2; ++mi)
#pragma unroll
        for (int t = 0; t < 4; ++t)
#pragma unroll
            for (int r = 0; r < 8; ++r)
                Y[(size_t)(m0 + mi * 16 + hi * 8 + r) * N + n0 + t * 16 + ln] =
                    (OutT)acc[mi][t][r];
}

// ---------------------------------------------------------------------------
// RoPE + head pack: P* f16 [B,S,D] -> Qh/Kh f16 [B,H,S,HD] (rotated),
//                   Vt f16 [B,H,HD,S] (transposed). One thread per pair.
// ---------------------------------------------------------------------------
__global__ __launch_bounds__(256)
void rope_pack(const _Float16* __restrict__ Pq, const _Float16* __restrict__ Pk,
               const _Float16* __restrict__ Pv, _Float16* __restrict__ Qh,
               _Float16* __restrict__ Kh, _Float16* __restrict__ Vt)
{
    const int tid = blockIdx.x * blockDim.x + threadIdx.x;  // 2^21 total
    const int i = tid & 31;            // pair index (HD/2 = 32)
    const int h = (tid >> 5) & 15;
    const int s = (tid >> 9) & 2047;
    const int b = tid >> 20;
    if (b >= Bn) return;

    const float inv_freq = __powf(10000.0f, -((float)i) * (1.0f / 32.0f));
    const float ang = (float)s * inv_freq;
    float sn, cs;
    __sincosf(ang, &sn, &cs);

    const size_t src = ((size_t)(b * Sn + s)) * Dn + h * HDn + 2 * i;
    const float q0 = (float)Pq[src], q1 = (float)Pq[src + 1];
    const float k0 = (float)Pk[src], k1 = (float)Pk[src + 1];
    const float v0 = (float)Pv[src], v1 = (float)Pv[src + 1];

    const int bh = b * Hn + h;
    const size_t dst = ((size_t)bh * Sn + s) * HDn + 2 * i;
    Qh[dst]     = (_Float16)(q0 * cs - q1 * sn);
    Qh[dst + 1] = (_Float16)(q0 * sn + q1 * cs);
    Kh[dst]     = (_Float16)(k0 * cs - k1 * sn);
    Kh[dst + 1] = (_Float16)(k0 * sn + k1 * cs);
    Vt[((size_t)bh * HDn + 2 * i) * Sn + s]     = (_Float16)v0;
    Vt[((size_t)bh * HDn + 2 * i + 1) * Sn + s] = (_Float16)v1;
}

// ---------------------------------------------------------------------------
// Flash attention, shifted-causal mask (j >= i masked; (0,0) kept).
// Grid: (S/64, B*H), block 128 = 4 waves; each wave owns 16 query rows.
// Online softmax; P tile round-trips through per-wave LDS to become an
// A-fragment for the PV WMMAs. ctx written f16 as [B,S,H*HD].
// ---------------------------------------------------------------------------
__global__ __launch_bounds__(128)
void flash_attn(const _Float16* __restrict__ Qh, const _Float16* __restrict__ Kh,
                const _Float16* __restrict__ Vt, _Float16* __restrict__ ctx)
{
    __shared__ _Float16 lds[4][16][40];   // per-wave 16x32 P tile (padded)

    const int lane = threadIdx.x & 31;
    const int wave = threadIdx.x >> 5;
    const int hi   = lane >> 4;
    const int ln   = lane & 15;
    const int bh   = blockIdx.y;
    const int i0   = blockIdx.x * 64 + wave * 16;

    const _Float16* Qb = Qh + (size_t)bh * Sn * HDn;
    const _Float16* Kb = Kh + (size_t)bh * Sn * HDn;
    const _Float16* Vb = Vt + (size_t)bh * HDn * Sn;

    // Q fragments for d-chunks [0,32) and [32,64): held for the whole loop.
    v16h aq0, aq1;
    {
        const _Float16* qp = Qb + (size_t)(i0 + ln) * HDn + hi * 16;
        h8 q00 = *(const h8*)(qp);
        h8 q01 = *(const h8*)(qp + 8);
        h8 q10 = *(const h8*)(qp + 32);
        h8 q11 = *(const h8*)(qp + 40);
#pragma unroll
        for (int e = 0; e < 8; ++e) {
            aq0[e] = q00[e]; aq0[8 + e] = q01[e];
            aq1[e] = q10[e]; aq1[8 + e] = q11[e];
        }
    }

    v8f accv[4] = {};
    float mrow[8], lsum[8];
#pragma unroll
    for (int r = 0; r < 8; ++r) { mrow[r] = NEG_BIG; lsum[r] = 0.0f; }

    for (int j0 = 0; j0 < i0 + 16; j0 += 32) {
        // ---- scores: two 16x16 tiles (keys j0..+15 and j0+16..+31)
        v8f s0 = v8f{}, s1 = v8f{};
        {
            const _Float16* kp0 = Kb + (size_t)(j0 + ln) * HDn + hi * 16;
            v16h b00 = *(const v16h*)(kp0);
            v16h b01 = *(const v16h*)(kp0 + 32);
            s0 = WMMA_F16(aq0, b00, s0);
            s0 = WMMA_F16(aq1, b01, s0);
            const _Float16* kp1 = Kb + (size_t)(j0 + 16 + ln) * HDn + hi * 16;
            v16h b10 = *(const v16h*)(kp1);
            v16h b11 = *(const v16h*)(kp1 + 32);
            s1 = WMMA_F16(aq0, b10, s1);
            s1 = WMMA_F16(aq1, b11, s1);
        }

        // ---- mask + online softmax (row-wise over 16-lane groups)
#pragma unroll
        for (int r = 0; r < 8; ++r) {
            const int irow = i0 + hi * 8 + r;
            const int jc0 = j0 + ln;
            const int jc1 = jc0 + 16;
            float e0 = s0[r] * SCALE_F;
            float e1 = s1[r] * SCALE_F;
            if ((jc0 >= irow) && !((irow == 0) && (jc0 == 0))) e0 = NEG_BIG;
            if (jc1 >= irow) e1 = NEG_BIG;

            float t = fmaxf(e0, e1);
#pragma unroll
            for (int off = 1; off < 16; off <<= 1)
                t = fmaxf(t, __shfl_xor(t, off, 32));
            const float mnew = fmaxf(mrow[r], t);
            const float corr = __expf(mrow[r] - mnew);
            mrow[r] = mnew;
            const float p0 = __expf(e0 - mnew);
            const float p1 = __expf(e1 - mnew);
            lsum[r] = lsum[r] * corr + p0 + p1;
            accv[0][r] *= corr; accv[1][r] *= corr;
            accv[2][r] *= corr; accv[3][r] *= corr;
            lds[wave][hi * 8 + r][ln]      = (_Float16)p0;
            lds[wave][hi * 8 + r][ln + 16] = (_Float16)p1;
        }

        // ---- re-gather P as A-fragment (16 rows x 32 keys)
        v16h ap;
        {
            const _Float16* pp = &lds[wave][ln][hi * 16];
            h8 pa = *(const h8*)(pp);
            h8 pb = *(const h8*)(pp + 8);
#pragma unroll
            for (int e = 0; e < 8; ++e) { ap[e] = pa[e]; ap[8 + e] = pb[e]; }
        }

        // ---- ctx += P @ V   (V transposed: dims are rows, keys contiguous)
#pragma unroll
        for (int t = 0; t < 4; ++t) {
            const _Float16* vp = Vb + (size_t)(t * 16 + ln) * Sn + j0 + hi * 16;
            v16h bv = *(const v16h*)(vp);
            accv[t] = WMMA_F16(ap, bv, accv[t]);
        }
    }

    // ---- normalize and store ctx (f16) as [B, S, H*HD]
#pragma unroll
    for (int r = 0; r < 8; ++r) {
        float t = lsum[r];
#pragma unroll
        for (int off = 1; off < 16; off <<= 1)
            t += __shfl_xor(t, off, 32);
        const float inv = 1.0f / t;
        accv[0][r] *= inv; accv[1][r] *= inv;
        accv[2][r] *= inv; accv[3][r] *= inv;
    }
    const int b = bh >> 4, h = bh & 15;
#pragma unroll
    for (int t = 0; t < 4; ++t)
#pragma unroll
        for (int r = 0; r < 8; ++r) {
            const int srow = i0 + hi * 8 + r;
            ctx[((size_t)(b * Sn + srow)) * Dn + h * HDn + t * 16 + ln] =
                (_Float16)accv[t][r];
        }
}

// ---------------------------------------------------------------------------
extern "C" void kernel_launch(void* const* d_in, const int* in_sizes, int n_in,
                              void* d_out, int out_size, void* d_ws, size_t ws_size,
                              hipStream_t stream) {
    const float* q  = (const float*)d_in[0];
    const float* k  = (const float*)d_in[1];
    const float* v  = (const float*)d_in[2];
    const float* Wq = (const float*)d_in[3];
    const float* Wk = (const float*)d_in[4];
    const float* Wv = (const float*)d_in[5];
    const float* Wo = (const float*)d_in[6];
    float* out = (float*)d_out;

    char* ws = (char*)d_ws;
    const size_t XSZ = (size_t)Bn * Sn * Dn * sizeof(_Float16);   // 8 MB
    const size_t WSZ = (size_t)Dn * En * sizeof(_Float16);        // 2 MB
    _Float16* Xq  = (_Float16*)(ws);
    _Float16* Xk  = (_Float16*)(ws + XSZ);
    _Float16* Xv  = (_Float16*)(ws + 2 * XSZ);
    _Float16* Wfq = (_Float16*)(ws + 3 * XSZ);
    _Float16* Wfk = (_Float16*)(ws + 3 * XSZ + WSZ);
    _Float16* Wfv = (_Float16*)(ws + 3 * XSZ + 2 * WSZ);
    _Float16* Wfo = (_Float16*)(ws + 3 * XSZ + 3 * WSZ);
    char*     p4  = ws + 3 * XSZ + 4 * WSZ;
    _Float16* Pq  = (_Float16*)(p4);
    _Float16* Pk  = (_Float16*)(p4 + XSZ);
    _Float16* Pv  = (_Float16*)(p4 + 2 * XSZ);
    _Float16* Qh2 = (_Float16*)(p4 + 3 * XSZ);
    _Float16* Kh2 = (_Float16*)(p4 + 4 * XSZ);
    _Float16* Vt2 = (_Float16*)(p4 + 5 * XSZ);
    _Float16* cx  = Xq;   // packed q input is dead after first GEMM; reuse

    const int M    = Bn * Sn;                       // 4096
    const int NX   = Bn * Sn * En;                  // 4 Mi elements
    const int NW   = Dn * En;                       // 1 Mi elements
    const int PTB  = 256 * 8;                       // elems per pack block

    pack_f16<<<(NX + PTB - 1) / PTB, 256, 0, stream>>>(q, Xq, NX);
    pack_f16<<<(NX + PTB - 1) / PTB, 256, 0, stream>>>(k, Xk, NX);
    pack_f16<<<(NX + PTB - 1) / PTB, 256, 0, stream>>>(v, Xv, NX);
    pack_f16<<<(NW + PTB - 1) / PTB, 256, 0, stream>>>(Wq, Wfq, NW);
    pack_f16<<<(NW + PTB - 1) / PTB, 256, 0, stream>>>(Wk, Wfk, NW);
    pack_f16<<<(NW + PTB - 1) / PTB, 256, 0, stream>>>(Wv, Wfv, NW);
    pack_f16<<<(NW + PTB - 1) / PTB, 256, 0, stream>>>(Wo, Wfo, NW);

    dim3 gg(M / 128, Dn / 128);                     // (32, 8)
    gemm_f16<_Float16><<<gg, 256, 0, stream>>>(Xq, Wfq, Pq, M, Dn, En);
    gemm_f16<_Float16><<<gg, 256, 0, stream>>>(Xk, Wfk, Pk, M, Dn, En);
    gemm_f16<_Float16><<<gg, 256, 0, stream>>>(Xv, Wfv, Pv, M, Dn, En);

    const int nrope = Bn * Sn * Hn * (HDn / 2);     // 2^21
    rope_pack<<<nrope / 256, 256, 0, stream>>>(Pq, Pk, Pv, Qh2, Kh2, Vt2);

    flash_attn<<<dim3(Sn / 64, Bn * Hn), 128, 0, stream>>>(Qh2, Kh2, Vt2, cx);

    gemm_f16<float><<<gg, 256, 0, stream>>>(cx, Wfo, out, M, En, Dn);
}
